// GraphConvProcessorBlock_27152783245686
// MI455X (gfx1250) — compile-verified, matrix operand order checked by reading, not run on hardware
//
#include <hip/hip_runtime.h>
#include <hip/hip_bf16.h>
#include <cstdint>

// ---------------------------------------------------------------------------
// GraphConv processor block for MI455X (gfx1250), f16 WMMA path.
//   d_out = [nodes_new (N*128 f32) | edges_new (E*128 f32)]
//   d_ws  = [agg (N*128 f32) | packed f16 weights]
// ---------------------------------------------------------------------------

#define D       128
#define NNODES  50000
#define NEDGES  600000

typedef __attribute__((ext_vector_type(16))) _Float16 v16h;
typedef __attribute__((ext_vector_type(8)))  float    v8f;

union FragU { uint4 u[2]; v16h h; };

__device__ __forceinline__ uint32_t pack2h(float a, float b) {
    union { _Float16 h[2]; uint32_t u; } p;
    p.h[0] = (_Float16)a; p.h[1] = (_Float16)b;
    return p.u;
}

// ---------------------------------------------------------------------------
// Weight packing: W is [K x 128] row-major f32. Output layout is the exact
// per-lane WMMA B-fragment stream: uint index = ((kblk*8 + nblk)*32 + lane)*8 + j
// holding f16 pair (K = kblk*32 + 16*(lane>=16) + 2j {,+1}, N = nblk*16 + lane%16).
// ---------------------------------------------------------------------------
__global__ void pack_weights(const float* __restrict__ W,
                             uint32_t* __restrict__ out, int K) {
    int t = blockIdx.x * 256 + threadIdx.x;
    if (t >= K * 64) return;
    int j    = t & 7;
    int lane = (t >> 3) & 31;
    int nblk = (t >> 8) & 7;
    int kblk = t >> 11;
    int n = nblk * 16 + (lane & 15);
    int k = kblk * 32 + ((lane >> 4) << 4) + 2 * j;
    out[t] = pack2h(W[(size_t)k * D + n], W[(size_t)(k + 1) * D + n]);
}

__global__ void zero_f32(float* __restrict__ p, int n) {
    int t = blockIdx.x * 256 + threadIdx.x;
    if (t < n) p[t] = 0.0f;
}

// ---------------------------------------------------------------------------
// Edge kernel: 16 edges per workgroup, 8 waves; wave w owns output cols
// [16w, 16w+16). GEMM1 K=384 (x_dst|x_src|edge_attr), SiLU, GEMM2 K=128,
// LayerNorm, write edges_new + atomic scatter-add into agg.
// ---------------------------------------------------------------------------
__global__ __launch_bounds__(256) void edge_kernel(
    const float* __restrict__ x, const float* __restrict__ edge_attr,
    const int* __restrict__ ei,
    const uint32_t* __restrict__ w1h, const float* __restrict__ b1,
    const uint32_t* __restrict__ w2h, const float* __restrict__ b2,
    const float* __restrict__ lng, const float* __restrict__ lnb,
    float* __restrict__ edges_out, float* __restrict__ agg)
{
    __shared__ __align__(16) uint32_t Alds[16 * 196];   // 16 x 384 f16 (pair-packed), padded stride
    __shared__ __align__(16) _Float16 Hlds[16 * 136];   // 16 x 128 f16, padded
    __shared__ __align__(16) float    Olds[16 * 132];   // 16 x 128 f32, padded
    __shared__ int dstL[16], srcL[16];

    const int tid  = threadIdx.x;
    const int lane = tid & 31;
    const int wave = tid >> 5;
    const int E0   = blockIdx.x * 16;

    if (tid < 16)       dstL[tid]      = ei[NEDGES + E0 + tid];
    else if (tid < 32)  srcL[tid - 16] = ei[E0 + tid - 16];
    __syncthreads();

    // Stage gathered inputs -> f16 LDS tile. Wave w fills rows 2w, 2w+1.
    for (int rr = 0; rr < 2; ++rr) {
        const int r = wave * 2 + rr;
        const float* seg0 = x + (size_t)dstL[r] * D;
        const float* seg1 = x + (size_t)srcL[r] * D;
        const float* seg2 = edge_attr + (size_t)(E0 + r) * D;
        const float* segs[3] = { seg0, seg1, seg2 };
        #pragma unroll
        for (int s = 0; s < 3; ++s)
            #pragma unroll
            for (int p = 0; p < 2; ++p) {
                const float2 f = *(const float2*)(segs[s] + p * 64 + lane * 2);
                Alds[r * 196 + s * 64 + p * 32 + lane] = pack2h(f.x, f.y);
            }
    }
    __syncthreads();

    const int m  = lane & 15;
    const int hi = lane >> 4;              // 0 or 1 (lane half)
    const int n  = wave * 16 + m;          // this lane's output column

    // ---- GEMM1: [16x384] * [384x128], 12 K-steps of 32 ----
    v8f acc = {};
    #pragma unroll
    for (int kb = 0; kb < 12; ++kb) {
        FragU A, B;
        const uint32_t* pa = &Alds[m * 196 + kb * 16 + hi * 8];
        A.u[0] = *(const uint4*)pa;
        A.u[1] = *(const uint4*)(pa + 4);
        const uint4* pb = (const uint4*)(w1h + (size_t)(((kb * 8 + wave) * 32 + lane) * 8));
        B.u[0] = pb[0];
        B.u[1] = pb[1];
        acc = __builtin_amdgcn_wmma_f32_16x16x32_f16(false, A.h, false, B.h,
                                                     (short)0, acc, false, false);
    }
    {
        const float bias = b1[n];
        #pragma unroll
        for (int r = 0; r < 8; ++r) {
            float v = acc[r] + bias;
            v = v / (1.0f + __expf(-v));                 // SiLU
            Hlds[(r + hi * 8) * 136 + n] = (_Float16)v;
        }
    }
    __syncthreads();

    // ---- GEMM2: [16x128] * [128x128], 4 K-steps ----
    v8f acc2 = {};
    #pragma unroll
    for (int kb = 0; kb < 4; ++kb) {
        FragU A, B;
        const uint32_t* pa = (const uint32_t*)Hlds + m * 68 + kb * 16 + hi * 8;
        A.u[0] = *(const uint4*)pa;
        A.u[1] = *(const uint4*)(pa + 4);
        const uint4* pb = (const uint4*)(w2h + (size_t)(((kb * 8 + wave) * 32 + lane) * 8));
        B.u[0] = pb[0];
        B.u[1] = pb[1];
        acc2 = __builtin_amdgcn_wmma_f32_16x16x32_f16(false, A.h, false, B.h,
                                                      (short)0, acc2, false, false);
    }
    {
        const float bias = b2[n];
        #pragma unroll
        for (int r = 0; r < 8; ++r)
            Olds[(r + hi * 8) * 132 + n] = acc2[r] + bias;
    }
    __syncthreads();

    // ---- LayerNorm per edge row + write + scatter-add. Wave w: rows 2w,2w+1.
    for (int rr = 0; rr < 2; ++rr) {
        const int r = wave * 2 + rr;
        const float4 v = *(const float4*)&Olds[r * 132 + lane * 4];
        float s  = v.x + v.y + v.z + v.w;
        float ss = v.x * v.x + v.y * v.y + v.z * v.z + v.w * v.w;
        #pragma unroll
        for (int off = 16; off; off >>= 1) {
            s  += __shfl_xor(s,  off);
            ss += __shfl_xor(ss, off);
        }
        const float mu   = s * (1.0f / 128.0f);
        const float rsig = rsqrtf(ss * (1.0f / 128.0f) - mu * mu + 1e-5f);
        const float o4[4] = { v.x, v.y, v.z, v.w };
        float* eo = edges_out + (size_t)(E0 + r) * D;
        float* ag = agg + (size_t)dstL[r] * D;
        #pragma unroll
        for (int c = 0; c < 4; ++c) {
            const int nn = lane * 4 + c;
            const float y = (o4[c] - mu) * rsig * lng[nn] + lnb[nn];
            eo[nn] = y;
            atomicAdd(&ag[nn], y);
        }
    }
}

// ---------------------------------------------------------------------------
// Node kernel: 16 nodes per workgroup. GEMM1 K=256 ([x|agg]), SiLU,
// GEMM2 K=128, LayerNorm, +residual, write nodes_new.
// ---------------------------------------------------------------------------
__global__ __launch_bounds__(256) void node_kernel(
    const float* __restrict__ x, const float* __restrict__ agg,
    const uint32_t* __restrict__ w1h, const float* __restrict__ b1,
    const uint32_t* __restrict__ w2h, const float* __restrict__ b2,
    const float* __restrict__ lng, const float* __restrict__ lnb,
    float* __restrict__ nodes_out)
{
    __shared__ __align__(16) uint32_t Alds[16 * 132];   // 16 x 256 f16 (pair-packed)
    __shared__ __align__(16) _Float16 Hlds[16 * 136];
    __shared__ __align__(16) float    Olds[16 * 132];

    const int tid  = threadIdx.x;
    const int lane = tid & 31;
    const int wave = tid >> 5;
    const int N0   = blockIdx.x * 16;

    for (int rr = 0; rr < 2; ++rr) {
        const int r = wave * 2 + rr;
        const float* seg0 = x   + (size_t)(N0 + r) * D;
        const float* seg1 = agg + (size_t)(N0 + r) * D;
        const float* segs[2] = { seg0, seg1 };
        #pragma unroll
        for (int s = 0; s < 2; ++s)
            #pragma unroll
            for (int p = 0; p < 2; ++p) {
                const float2 f = *(const float2*)(segs[s] + p * 64 + lane * 2);
                Alds[r * 132 + s * 64 + p * 32 + lane] = pack2h(f.x, f.y);
            }
    }
    __syncthreads();

    const int m  = lane & 15;
    const int hi = lane >> 4;
    const int n  = wave * 16 + m;

    v8f acc = {};
    #pragma unroll
    for (int kb = 0; kb < 8; ++kb) {
        FragU A, B;
        const uint32_t* pa = &Alds[m * 132 + kb * 16 + hi * 8];
        A.u[0] = *(const uint4*)pa;
        A.u[1] = *(const uint4*)(pa + 4);
        const uint4* pb = (const uint4*)(w1h + (size_t)(((kb * 8 + wave) * 32 + lane) * 8));
        B.u[0] = pb[0];
        B.u[1] = pb[1];
        acc = __builtin_amdgcn_wmma_f32_16x16x32_f16(false, A.h, false, B.h,
                                                     (short)0, acc, false, false);
    }
    {
        const float bias = b1[n];
        #pragma unroll
        for (int r = 0; r < 8; ++r) {
            float v = acc[r] + bias;
            v = v / (1.0f + __expf(-v));
            Hlds[(r + hi * 8) * 136 + n] = (_Float16)v;
        }
    }
    __syncthreads();

    v8f acc2 = {};
    #pragma unroll
    for (int kb = 0; kb < 4; ++kb) {
        FragU A, B;
        const uint32_t* pa = (const uint32_t*)Hlds + m * 68 + kb * 16 + hi * 8;
        A.u[0] = *(const uint4*)pa;
        A.u[1] = *(const uint4*)(pa + 4);
        const uint4* pb = (const uint4*)(w2h + (size_t)(((kb * 8 + wave) * 32 + lane) * 8));
        B.u[0] = pb[0];
        B.u[1] = pb[1];
        acc2 = __builtin_amdgcn_wmma_f32_16x16x32_f16(false, A.h, false, B.h,
                                                      (short)0, acc2, false, false);
    }
    {
        const float bias = b2[n];
        #pragma unroll
        for (int r = 0; r < 8; ++r)
            Olds[(r + hi * 8) * 132 + n] = acc2[r] + bias;
    }
    __syncthreads();

    for (int rr = 0; rr < 2; ++rr) {
        const int r = wave * 2 + rr;
        const int node = N0 + r;
        const float4 v = *(const float4*)&Olds[r * 132 + lane * 4];
        float s  = v.x + v.y + v.z + v.w;
        float ss = v.x * v.x + v.y * v.y + v.z * v.z + v.w * v.w;
        #pragma unroll
        for (int off = 16; off; off >>= 1) {
            s  += __shfl_xor(s,  off);
            ss += __shfl_xor(ss, off);
        }
        const float mu   = s * (1.0f / 128.0f);
        const float rsig = rsqrtf(ss * (1.0f / 128.0f) - mu * mu + 1e-5f);
        const float o4[4] = { v.x, v.y, v.z, v.w };
        const float* xr = x + (size_t)node * D;
        float* no = nodes_out + (size_t)node * D;
        #pragma unroll
        for (int c = 0; c < 4; ++c) {
            const int nn = lane * 4 + c;
            no[nn] = (o4[c] - mu) * rsig * lng[nn] + lnb[nn] + xr[nn];
        }
    }
}

// ---------------------------------------------------------------------------
extern "C" void kernel_launch(void* const* d_in, const int* in_sizes, int n_in,
                              void* d_out, int out_size, void* d_ws, size_t ws_size,
                              hipStream_t stream)
{
    (void)in_sizes; (void)n_in; (void)out_size; (void)ws_size;

    const float* x    = (const float*)d_in[0];
    const float* ea   = (const float*)d_in[1];
    const int*   ei   = (const int*)d_in[2];
    const float* eW1  = (const float*)d_in[3];
    const float* eb1  = (const float*)d_in[4];
    const float* eW2  = (const float*)d_in[5];
    const float* eb2  = (const float*)d_in[6];
    const float* elng = (const float*)d_in[7];
    const float* elnb = (const float*)d_in[8];
    const float* nW1  = (const float*)d_in[9];
    const float* nb1  = (const float*)d_in[10];
    const float* nW2  = (const float*)d_in[11];
    const float* nb2  = (const float*)d_in[12];
    const float* nlng = (const float*)d_in[13];
    const float* nlnb = (const float*)d_in[14];

    float* nodes_out = (float*)d_out;
    float* edges_out = (float*)d_out + (size_t)NNODES * D;

    float*    agg  = (float*)d_ws;
    uint32_t* w1h  = (uint32_t*)(agg + (size_t)NNODES * D);   // 384*64 uints
    uint32_t* w2h  = w1h  + 384 * 64;                         // 128*64
    uint32_t* nw1h = w2h  + 128 * 64;                         // 256*64
    uint32_t* nw2h = nw1h + 256 * 64;                         // 128*64

    const int aggN = NNODES * D;
    zero_f32<<<(aggN + 255) / 256, 256, 0, stream>>>(agg, aggN);
    pack_weights<<<(384 * 64 + 255) / 256, 256, 0, stream>>>(eW1, w1h, 384);
    pack_weights<<<(128 * 64 + 255) / 256, 256, 0, stream>>>(eW2, w2h, 128);
    pack_weights<<<(256 * 64 + 255) / 256, 256, 0, stream>>>(nW1, nw1h, 256);
    pack_weights<<<(128 * 64 + 255) / 256, 256, 0, stream>>>(nW2, nw2h, 128);

    edge_kernel<<<NEDGES / 16, 256, 0, stream>>>(x, ea, ei, w1h, eb1, w2h, eb2,
                                                 elng, elnb, edges_out, agg);
    node_kernel<<<NNODES / 16, 256, 0, stream>>>(x, agg, nw1h, nb1, nw2h, nb2,
                                                 nlng, nlnb, nodes_out);
}
